// GConvGRU_19473381720232
// MI455X (gfx1250) — matrix operand order repaired and checked by reading, not compiled, not run on hardware
//
#include <hip/hip_runtime.h>
#include <math.h>

#define DEV __device__ __forceinline__

static constexpr int Nn = 20000;   // nodes
static constexpr int Ne = 320000;  // edges
static constexpr int Dd = 512;     // feature dim (DIN == DOUT)

typedef __attribute__((ext_vector_type(16))) __bf16       v16bf;
typedef __attribute__((ext_vector_type(8)))  float        v8f;
typedef __attribute__((ext_vector_type(4)))  unsigned int u32x4;

union Frag { v16bf v; u32x4 u[2]; };

DEV unsigned short f2bf(float f) {
  union { float f; unsigned int u; } c; c.f = f;
  unsigned int u = c.u;
  if ((u & 0x7fffffffu) > 0x7f800000u) return (unsigned short)((u >> 16) | 0x40u); // quiet NaN
  return (unsigned short)((u + 0x7fffu + ((u >> 16) & 1u)) >> 16);                 // RNE
}

DEV void atomAddF(float* p, float v) {
  __hip_atomic_fetch_add(p, v, __ATOMIC_RELAXED, __HIP_MEMORY_SCOPE_AGENT);
}
DEV int atomAddI(int* p, int v) {
  return __hip_atomic_fetch_add(p, v, __ATOMIC_RELAXED, __HIP_MEMORY_SCOPE_AGENT);
}

struct f4 { float x, y, z, w; };
DEV f4 ld4(const float* p) { float4 t = *(const float4*)p; return {t.x, t.y, t.z, t.w}; }
DEV void fma4(f4& a, float w, const f4& b) {
  a.x = fmaf(w, b.x, a.x); a.y = fmaf(w, b.y, a.y);
  a.z = fmaf(w, b.z, a.z); a.w = fmaf(w, b.w, a.w);
}
DEV void st4bf(unsigned short* p, const f4& a) {
  ushort4 o; o.x = f2bf(a.x); o.y = f2bf(a.y); o.z = f2bf(a.z); o.w = f2bf(a.w);
  *(ushort4*)p = o;
}

// ---------------- graph-norm + CSR build ----------------

__global__ void k_init(float* __restrict__ deg, int* __restrict__ cnt) {
  int i = blockIdx.x * blockDim.x + threadIdx.x;
  if (i < Nn) { deg[i] = 1.0f; cnt[i] = 0; }  // self-loop weight; in-degree counter
}

__global__ void k_deg_cnt(const int* __restrict__ cols, const float* __restrict__ ew,
                          float* __restrict__ deg, int* __restrict__ cnt) {
  int e = blockIdx.x * blockDim.x + threadIdx.x;
  if (e < Ne) {
    int c = cols[e];
    atomAddF(&deg[c], ew[e]);
    atomAddI(&cnt[c], 1);
  }
}

__global__ void k_dinv(float* __restrict__ deg, float* __restrict__ selfnorm) {
  int i = blockIdx.x * blockDim.x + threadIdx.x;
  if (i < Nn) {
    float d = deg[i];
    float dv = (d > 0.0f) ? rsqrtf(d) : 0.0f;
    deg[i] = dv;              // in-place: deg buffer becomes dinv
    selfnorm[i] = dv * dv;    // self-loop edge norm (weight 1)
  }
}

__global__ void k_edge_norm(const int* __restrict__ rows, const int* __restrict__ cols,
                            const float* __restrict__ ew, const float* __restrict__ dinv,
                            float* __restrict__ norm) {
  int e = blockIdx.x * blockDim.x + threadIdx.x;
  if (e < Ne) norm[e] = dinv[rows[e]] * ew[e] * dinv[cols[e]];
}

// single-block exclusive prefix sum over cnt[0..Nn) -> rowptr, cursor (LDS scan)
__global__ __launch_bounds__(1024) void k_scan(const int* __restrict__ cnt,
                                               int* __restrict__ rowptr,
                                               int* __restrict__ cursor) {
  __shared__ int sdata[1024];
  __shared__ int running;
  const int tid = threadIdx.x;
  if (tid == 0) running = 0;
  __syncthreads();
  for (int base = 0; base < Nn; base += 1024) {
    int i = base + tid;
    int v = (i < Nn) ? cnt[i] : 0;
    sdata[tid] = v;
    __syncthreads();
    // Hillis-Steele inclusive scan
    for (int offs = 1; offs < 1024; offs <<= 1) {
      int t = (tid >= offs) ? sdata[tid - offs] : 0;
      __syncthreads();
      sdata[tid] += t;
      __syncthreads();
    }
    int total = sdata[1023];
    int excl  = sdata[tid] - v;
    int r = running;
    if (i < Nn) { rowptr[i] = r + excl; cursor[i] = r + excl; }
    __syncthreads();                    // everyone has consumed `running`
    if (tid == 0) running = r + total;
    __syncthreads();
  }
  if (tid == 0) rowptr[Nn] = running;   // == Ne
}

__global__ void k_fill(const int* __restrict__ cols, int* __restrict__ cursor,
                       int* __restrict__ eid) {
  int e = blockIdx.x * blockDim.x + threadIdx.x;
  if (e < Ne) {
    int pos = atomAddI(&cursor[cols[e]], 1);
    eid[pos] = e;
  }
}

// ---------------- CSR gather: dst[i] = bf16( snorm[i]*m(i) + sum_e norm*m(row) )
// HRMODE=0: two independent sources (X,H) -> two bf16 outputs
// HRMODE=1: message = src0 * src1 elementwise (H*R) -> one bf16 output
template <int HRMODE>
__global__ __launch_bounds__(128) void k_gather(
    const int* __restrict__ rowptr, const int* __restrict__ eid,
    const int* __restrict__ rows, const float* __restrict__ norm,
    const float* __restrict__ snorm,
    const float* __restrict__ src0, const float* __restrict__ src1,
    unsigned short* __restrict__ dst0, unsigned short* __restrict__ dst1) {
  const int i = blockIdx.x;             // destination node
  const int f = threadIdx.x << 2;       // 128 threads * float4 = 512 features
  const float s = snorm[i];

  f4 a0, a1;
  {
    f4 x = ld4(src0 + (size_t)i * Dd + f);
    f4 h = ld4(src1 + (size_t)i * Dd + f);
    if (HRMODE) {
      a0 = {s * x.x * h.x, s * x.y * h.y, s * x.z * h.z, s * x.w * h.w};
      a1 = {0, 0, 0, 0};
    } else {
      a0 = {s * x.x, s * x.y, s * x.z, s * x.w};
      a1 = {s * h.x, s * h.y, s * h.z, s * h.w};
    }
  }

  const int pEnd = rowptr[i + 1];
  for (int p = rowptr[i]; p < pEnd; ++p) {
    int e = eid[p];
    int r = rows[e];
    float w = norm[e];
    f4 x = ld4(src0 + (size_t)r * Dd + f);
    if (HRMODE) {
      f4 h = ld4(src1 + (size_t)r * Dd + f);
      f4 m = {x.x * h.x, x.y * h.y, x.z * h.z, x.w * h.w};
      fma4(a0, w, m);
    } else {
      fma4(a0, w, x);
      f4 h = ld4(src1 + (size_t)r * Dd + f);
      fma4(a1, w, h);
    }
  }

  st4bf(dst0 + (size_t)i * Dd + f, a0);
  if (!HRMODE) st4bf(dst1 + (size_t)i * Dd + f, a1);
}

// ---------------- weight prep: W [k,n] f32 -> Wt [n,k] bf16 ----------------

__global__ void k_prep_w(const float* __restrict__ W, unsigned short* __restrict__ Wt) {
  int idx = blockIdx.x * blockDim.x + threadIdx.x;  // over Dd*Dd
  if (idx >= Dd * Dd) return;
  int n = idx & (Dd - 1);
  int k = idx >> 9;
  Wt[(size_t)n * Dd + k] = f2bf(W[(size_t)k * Dd + n]);
}

// ---------------- WMMA dual-GEMM + fused epilogue ----------------
// Wave computes 16 rows x 64 cols of  act(A0@W0t + A1@W1t + b0 + b1).
// FINAL=0: sigmoid -> store to `out` (gate).
// FINAL=1: tanh -> out = Z*H + (1-Z)*tanh(.)  (GRU combine).
template <int FINAL>
__global__ __launch_bounds__(256) void k_gemm_dual(
    const unsigned short* __restrict__ A0, const unsigned short* __restrict__ W0t,
    const float* __restrict__ b0,
    const unsigned short* __restrict__ A1, const unsigned short* __restrict__ W1t,
    const float* __restrict__ b1,
    const float* __restrict__ Z, const float* __restrict__ H,
    float* __restrict__ out) {
  const int lane = threadIdx.x & 31;
  const int wave = threadIdx.x >> 5;      // 0..7 -> 64-col slab
  const int row0 = blockIdx.x << 4;       // 16 rows per block (Nn/16 = 1250 blocks)
  const int col0 = wave << 6;
  const int m  = lane & 15;
  const int hi = lane >> 4;               // lane half selects K sub-range

  const unsigned short* Abase[2] = {A0 + (size_t)(row0 + m) * Dd + hi * 8,
                                    A1 + (size_t)(row0 + m) * Dd + hi * 8};
  const unsigned short* Wbase[2] = {W0t + (size_t)(col0 + m) * Dd + (hi << 3),
                                    W1t + (size_t)(col0 + m) * Dd + (hi << 3)};

  const v8f vzero = {0.f, 0.f, 0.f, 0.f, 0.f, 0.f, 0.f, 0.f};
  v8f acc[4] = {vzero, vzero, vzero, vzero};

  Frag a_c, a_n;
  Frag b_c[4], b_n[4];

  auto loadStep = [&](Frag& a, Frag* b, int s) {
    const int p  = s >> 4;           // which of the two GEMMs
    const int k0 = (s & 15) << 5;    // K offset within that GEMM
    const unsigned short* ar = Abase[p] + k0;
    a.u[0] = *(const u32x4*)(ar);
    a.u[1] = *(const u32x4*)(ar + 16);
#pragma unroll
    for (int j = 0; j < 4; ++j) {
      const unsigned short* br = Wbase[p] + (size_t)(j * 16) * Dd + k0;
      b[j].u[0] = *(const u32x4*)(br);
      b[j].u[1] = *(const u32x4*)(br + 16);
    }
  };

  loadStep(a_c, b_c, 0);
#pragma unroll
  for (int s = 0; s < 32; ++s) {
    if (s + 1 < 32) loadStep(a_n, b_n, s + 1);  // prefetch next K-step
#pragma unroll
    for (int j = 0; j < 4; ++j)
      acc[j] = __builtin_amdgcn_wmma_f32_16x16x32_bf16(
          false, a_c.v, false, b_c[j].v, (short)0, acc[j], false, false);
    if (s + 1 < 32) {
      a_c = a_n;
#pragma unroll
      for (int j = 0; j < 4; ++j) b_c[j] = b_n[j];
    }
  }

#pragma unroll
  for (int j = 0; j < 4; ++j) {
    int c = col0 + j * 16 + m;
    float bias = b0[c] + b1[c];
#pragma unroll
    for (int v = 0; v < 8; ++v) {
      int r = row0 + v + hi * 8;          // C/D layout: VGPR v -> row v (+8 for high lanes)
      size_t idx = (size_t)r * Dd + c;
      float x = acc[j][v] + bias;
      if (FINAL) {
        float t = tanhf(x);
        float z = Z[idx];
        out[idx] = z * H[idx] + (1.0f - z) * t;
      } else {
        out[idx] = 1.0f / (1.0f + __expf(-x));
      }
    }
  }
}

// ---------------- host launch ----------------

extern "C" void kernel_launch(void* const* d_in, const int* in_sizes, int n_in,
                              void* d_out, int out_size, void* d_ws, size_t ws_size,
                              hipStream_t stream) {
  (void)in_sizes; (void)n_in; (void)out_size; (void)ws_size;
  const float* X    = (const float*)d_in[0];
  const int*   ei   = (const int*)d_in[1];
  const float* ew   = (const float*)d_in[2];
  const float* H    = (const float*)d_in[3];
  const float* W_xz = (const float*)d_in[4];  const float* b_xz = (const float*)d_in[5];
  const float* W_hz = (const float*)d_in[6];  const float* b_hz = (const float*)d_in[7];
  const float* W_xr = (const float*)d_in[8];  const float* b_xr = (const float*)d_in[9];
  const float* W_hr = (const float*)d_in[10]; const float* b_hr = (const float*)d_in[11];
  const float* W_xh = (const float*)d_in[12]; const float* b_xh = (const float*)d_in[13];
  const float* W_hh = (const float*)d_in[14]; const float* b_hh = (const float*)d_in[15];
  float* out = (float*)d_out;

  const int* rows = ei;       // edge_index[0]
  const int* cols = ei + Ne;  // edge_index[1]

  // workspace carve-up
  char* base = (char*)d_ws;
  size_t off = 0;
  auto take = [&](size_t bytes) -> void* {
    void* p = base + off;
    off = (off + bytes + 255) & ~(size_t)255;
    return p;
  };
  const size_t nodeF = (size_t)Nn * Dd;
  float*          deg    = (float*)take((size_t)Nn * 4);       // becomes dinv
  float*          snorm  = (float*)take((size_t)Nn * 4);
  float*          enorm  = (float*)take((size_t)Ne * 4);
  int*            cnt    = (int*)take((size_t)Nn * 4);
  int*            rowptr = (int*)take((size_t)(Nn + 1) * 4);
  int*            cursor = (int*)take((size_t)Nn * 4);
  int*            eid    = (int*)take((size_t)Ne * 4);
  unsigned short* SXb    = (unsigned short*)take(nodeF * 2);   // bf16 S@X
  unsigned short* SHb    = (unsigned short*)take(nodeF * 2);   // bf16 S@H, reused for S@(H*R)
  float*          Zbuf   = (float*)take(nodeF * 4);
  float*          Rbuf   = (float*)take(nodeF * 4);
  unsigned short* Wt[6];
  const float* Wsrc[6] = {W_xz, W_hz, W_xr, W_hr, W_xh, W_hh};
  for (int i = 0; i < 6; ++i) Wt[i] = (unsigned short*)take((size_t)Dd * Dd * 2);

  const int nThr = 256;
  const int nodeBlocks = (Nn + nThr - 1) / nThr;
  const int edgeBlocks = (Ne + nThr - 1) / nThr;
  const int gemmBlocks = Nn / 16;  // 1250

  // weights: transpose + bf16
  for (int i = 0; i < 6; ++i)
    k_prep_w<<<(Dd * Dd + nThr - 1) / nThr, nThr, 0, stream>>>(Wsrc[i], Wt[i]);

  // gcn_norm + CSR build (by destination)
  k_init<<<nodeBlocks, nThr, 0, stream>>>(deg, cnt);
  k_deg_cnt<<<edgeBlocks, nThr, 0, stream>>>(cols, ew, deg, cnt);
  k_dinv<<<nodeBlocks, nThr, 0, stream>>>(deg, snorm);
  k_edge_norm<<<edgeBlocks, nThr, 0, stream>>>(rows, cols, ew, deg, enorm);
  k_scan<<<1, 1024, 0, stream>>>(cnt, rowptr, cursor);
  k_fill<<<edgeBlocks, nThr, 0, stream>>>(cols, cursor, eid);

  // SX = S@X, SH = S@H  (atomic-free register gather, direct bf16 output)
  k_gather<0><<<Nn, 128, 0, stream>>>(rowptr, eid, rows, enorm, snorm, X, H, SXb, SHb);

  // Z = sigmoid(SX@Wxz + SH@Whz + b), R = sigmoid(SX@Wxr + SH@Whr + b)
  k_gemm_dual<0><<<gemmBlocks, 256, 0, stream>>>(SXb, Wt[0], b_xz, SHb, Wt[1], b_hz,
                                                 nullptr, nullptr, Zbuf);
  k_gemm_dual<0><<<gemmBlocks, 256, 0, stream>>>(SXb, Wt[2], b_xr, SHb, Wt[3], b_hr,
                                                 nullptr, nullptr, Rbuf);

  // S@(H*R) computed on the fly from H and R (reuse SHb)
  k_gather<1><<<Nn, 128, 0, stream>>>(rowptr, eid, rows, enorm, snorm, H, Rbuf, SHb, nullptr);

  // H_tilde = tanh(SX@Wxh + S(H*R)@Whh + b); out = Z*H + (1-Z)*H_tilde
  k_gemm_dual<1><<<gemmBlocks, 256, 0, stream>>>(SXb, Wt[4], b_xh, SHb, Wt[5], b_hh,
                                                 Zbuf, H, out);
}
